// InnovativeSRDELayer_67061619360305
// MI455X (gfx1250) — matrix operand order repaired and checked by reading, not compiled, not run on hardware
//
#include <hip/hip_runtime.h>
#include <hip/hip_bf16.h>
#include <math.h>

// ---------------- problem constants (match reference) ----------------
constexpr int Bc  = 2;
constexpr int Sc  = 2048;
constexpr int Dc  = 1024;
constexpr int Ec  = 8;
constexpr int Cc  = 64;
constexpr int Mc  = 128;      // memory slots
constexpr int Zc  = 3;
constexpr int Tc  = Bc * Sc;  // 4096 tokens
constexpr int D4c = 4 * Dc;   // 4096
constexpr int ZEc = Zc * Ec;  // 24
constexpr int DGc = Dc / 4;   // 256

typedef __attribute__((ext_vector_type(16))) __bf16 v16bf;
typedef __attribute__((ext_vector_type(8)))  float  v8f;

// ---------------- helpers ----------------
static __device__ __forceinline__ unsigned short f32_to_bf16(float f) {
  union { float f; unsigned int u; } v; v.f = f;
  unsigned int u = v.u;
  return (unsigned short)((u + 0x7FFFu + ((u >> 16) & 1u)) >> 16);
}
static __device__ __forceinline__ float bf16_to_f32(unsigned short h) {
  union { unsigned int u; float f; } v; v.u = ((unsigned int)h) << 16; return v.f;
}
static __device__ __forceinline__ float gelu_exact(float x) {
  return 0.5f * x * (1.0f + erff(x * 0.7071067811865475f));
}

// ---------------- bf16 WMMA GEMM: C[M,N] = act(alpha*A@B + bias) ----------------
// A: MxK bf16 row-major. B: KxN bf16 row-major (BTRANS=0) or NxK row-major (BTRANS=1).
// Block: 256 threads (8 waves, 4x2). Block tile 256x128, wave tile 64x64 (4x4 WMMA),
// K-step 32. Next K-tile is register-prefetched (named scalars, no arrays -> no
// scratch spills) while the 16 WMMAs execute. Requires M%256==0, N%128==0, K%32==0.
#define TM 256
#define TN 128
#define TK 32

template <int BTRANS>
__global__ __launch_bounds__(256) __attribute__((amdgpu_waves_per_eu(1)))
void gemm_bf16_wmma(
    const unsigned short* __restrict__ A, const unsigned short* __restrict__ Bm,
    int M, int N, int K,
    const float* __restrict__ bias, float alpha, int act,
    const float* __restrict__ row_scale, int rs_stride, int accum,
    float* __restrict__ outF, unsigned short* __restrict__ outB)
{
  __shared__ __align__(32) unsigned short ldsA[TM * TK];   // [row][k]
  __shared__ __align__(32) unsigned short ldsB[TN * TK];   // [n][k]

  const int tid   = threadIdx.x;
  const int lane  = tid & 31;
  const int wave  = tid >> 5;
  const int waveM = wave >> 1;                // 0..3 -> 64 rows each
  const int waveN = wave & 1;                 // 0..1 -> 64 cols each
  const size_t blockM = (size_t)blockIdx.y * TM;
  const size_t blockN = (size_t)blockIdx.x * TN;

  v8f acc00 = (v8f)0.0f, acc01 = (v8f)0.0f, acc02 = (v8f)0.0f, acc03 = (v8f)0.0f;
  v8f acc10 = (v8f)0.0f, acc11 = (v8f)0.0f, acc12 = (v8f)0.0f, acc13 = (v8f)0.0f;
  v8f acc20 = (v8f)0.0f, acc21 = (v8f)0.0f, acc22 = (v8f)0.0f, acc23 = (v8f)0.0f;
  v8f acc30 = (v8f)0.0f, acc31 = (v8f)0.0f, acc32 = (v8f)0.0f, acc33 = (v8f)0.0f;

  const int koff = (lane >= 16) ? 16 : 0;     // A frag: K in {koff..koff+15}
  const int kb   = (lane >= 16) ? 16 : 0;     // B frag: K in {kb..kb+15}
  const int ln15 = lane & 15;

  // ---- A staging: chunk c = tid + 256*i -> row = (tid>>2)+64*i, col = (tid&3)*8 ----
  const int aRow = tid >> 2;
  const int aCol = (tid & 3) * 8;
  const unsigned short* aG0 = A + (blockM + aRow) * K + aCol;
  const unsigned short* aG1 = aG0 + (size_t)64 * K;
  const unsigned short* aG2 = aG0 + (size_t)128 * K;
  const unsigned short* aG3 = aG0 + (size_t)192 * K;
  unsigned short* aS = &ldsA[aRow * TK + aCol];            // +2048*i per chunk

  // ---- B staging ----
  // BTRANS=1: chunk c -> n=(tid>>2)+64*i, kOff=(tid&3)*8 (contiguous in k)
  // BTRANS=0: chunk c -> k=(tid>>4)+16*i, nOff=(tid&15)*8 (scatter-transpose to [n][k])
  const int bN    = tid >> 2;
  const int bKoff = (tid & 3) * 8;
  const int bK    = tid >> 4;
  const int bNoff = (tid & 15) * 8;
  const unsigned short* bG0;
  const unsigned short* bG1;
  unsigned short* bS;
  if (BTRANS) {
    bG0 = Bm + (blockN + bN) * K + bKoff;
    bG1 = bG0 + (size_t)64 * K;
    bS  = &ldsB[bN * TK + bKoff];                          // +2048 for chunk 1
  } else {
    bG0 = Bm + (size_t)bK * N + blockN + bNoff;
    bG1 = bG0 + (size_t)16 * N;
    bS  = &ldsB[bNoff * TK + bK];                          // +e*TK scatter; +16 for chunk 1
  }

  // ---- fragment read pointers (constant per thread) ----
  const unsigned short* aF = &ldsA[(waveM * 64 + ln15) * TK + koff];  // +512*i per row tile
  const unsigned short* bF = &ldsB[(waveN * 64 + ln15) * TK + kb];    // +512*j per col tile

  uint4 pa0, pa1, pa2, pa3, pb0, pb1;
  // prologue: load K-tile 0
  pa0 = *reinterpret_cast<const uint4*>(aG0);
  pa1 = *reinterpret_cast<const uint4*>(aG1);
  pa2 = *reinterpret_cast<const uint4*>(aG2);
  pa3 = *reinterpret_cast<const uint4*>(aG3);
  pb0 = *reinterpret_cast<const uint4*>(bG0);
  pb1 = *reinterpret_cast<const uint4*>(bG1);
  aG0 += TK; aG1 += TK; aG2 += TK; aG3 += TK;
  if (BTRANS) { bG0 += TK; bG1 += TK; }
  else        { bG0 += (size_t)TK * N; bG1 += (size_t)TK * N; }

  const int nkt = K / TK;
  for (int kt = 0; kt < nkt; ++kt) {
    __syncthreads();            // previous tiles fully consumed
    *reinterpret_cast<uint4*>(aS)         = pa0;
    *reinterpret_cast<uint4*>(aS + 2048)  = pa1;
    *reinterpret_cast<uint4*>(aS + 4096)  = pa2;
    *reinterpret_cast<uint4*>(aS + 6144)  = pa3;
    if (BTRANS) {
      *reinterpret_cast<uint4*>(bS)        = pb0;
      *reinterpret_cast<uint4*>(bS + 2048) = pb1;
    } else {
      union { uint4 q; unsigned short s[8]; } u0, u1;
      u0.q = pb0; u1.q = pb1;
#pragma unroll
      for (int e = 0; e < 8; ++e) bS[e * TK]      = u0.s[e];
#pragma unroll
      for (int e = 0; e < 8; ++e) bS[e * TK + 16] = u1.s[e];
    }
    __syncthreads();            // tiles ready

    if (kt + 1 < nkt) {         // next-tile loads fly during the WMMAs below
      pa0 = *reinterpret_cast<const uint4*>(aG0);
      pa1 = *reinterpret_cast<const uint4*>(aG1);
      pa2 = *reinterpret_cast<const uint4*>(aG2);
      pa3 = *reinterpret_cast<const uint4*>(aG3);
      pb0 = *reinterpret_cast<const uint4*>(bG0);
      pb1 = *reinterpret_cast<const uint4*>(bG1);
      aG0 += TK; aG1 += TK; aG2 += TK; aG3 += TK;
      if (BTRANS) { bG0 += TK; bG1 += TK; }
      else        { bG0 += (size_t)TK * N; bG1 += (size_t)TK * N; }
    }

    // ---- B fragments (32 VGPRs) ----
    v16bf b0 = *reinterpret_cast<const v16bf*>(bF);
    v16bf b1 = *reinterpret_cast<const v16bf*>(bF + 512);
    v16bf b2 = *reinterpret_cast<const v16bf*>(bF + 1024);
    v16bf b3 = *reinterpret_cast<const v16bf*>(bF + 1536);

    // ---- per-row A fragment + 4 WMMAs each ----
    union { uint4 q[2]; v16bf v; } fa;
    fa.q[0] = *reinterpret_cast<const uint4*>(aF);
    fa.q[1] = *reinterpret_cast<const uint4*>(aF + 8);
    acc00 = __builtin_amdgcn_wmma_f32_16x16x32_bf16(false, fa.v, false, b0, (short)0, acc00, false, false);
    acc01 = __builtin_amdgcn_wmma_f32_16x16x32_bf16(false, fa.v, false, b1, (short)0, acc01, false, false);
    acc02 = __builtin_amdgcn_wmma_f32_16x16x32_bf16(false, fa.v, false, b2, (short)0, acc02, false, false);
    acc03 = __builtin_amdgcn_wmma_f32_16x16x32_bf16(false, fa.v, false, b3, (short)0, acc03, false, false);
    fa.q[0] = *reinterpret_cast<const uint4*>(aF + 512);
    fa.q[1] = *reinterpret_cast<const uint4*>(aF + 520);
    acc10 = __builtin_amdgcn_wmma_f32_16x16x32_bf16(false, fa.v, false, b0, (short)0, acc10, false, false);
    acc11 = __builtin_amdgcn_wmma_f32_16x16x32_bf16(false, fa.v, false, b1, (short)0, acc11, false, false);
    acc12 = __builtin_amdgcn_wmma_f32_16x16x32_bf16(false, fa.v, false, b2, (short)0, acc12, false, false);
    acc13 = __builtin_amdgcn_wmma_f32_16x16x32_bf16(false, fa.v, false, b3, (short)0, acc13, false, false);
    fa.q[0] = *reinterpret_cast<const uint4*>(aF + 1024);
    fa.q[1] = *reinterpret_cast<const uint4*>(aF + 1032);
    acc20 = __builtin_amdgcn_wmma_f32_16x16x32_bf16(false, fa.v, false, b0, (short)0, acc20, false, false);
    acc21 = __builtin_amdgcn_wmma_f32_16x16x32_bf16(false, fa.v, false, b1, (short)0, acc21, false, false);
    acc22 = __builtin_amdgcn_wmma_f32_16x16x32_bf16(false, fa.v, false, b2, (short)0, acc22, false, false);
    acc23 = __builtin_amdgcn_wmma_f32_16x16x32_bf16(false, fa.v, false, b3, (short)0, acc23, false, false);
    fa.q[0] = *reinterpret_cast<const uint4*>(aF + 1536);
    fa.q[1] = *reinterpret_cast<const uint4*>(aF + 1544);
    acc30 = __builtin_amdgcn_wmma_f32_16x16x32_bf16(false, fa.v, false, b0, (short)0, acc30, false, false);
    acc31 = __builtin_amdgcn_wmma_f32_16x16x32_bf16(false, fa.v, false, b1, (short)0, acc31, false, false);
    acc32 = __builtin_amdgcn_wmma_f32_16x16x32_bf16(false, fa.v, false, b2, (short)0, acc32, false, false);
    acc33 = __builtin_amdgcn_wmma_f32_16x16x32_bf16(false, fa.v, false, b3, (short)0, acc33, false, false);
  }

  // ---- epilogue: VGPR r of 16x16 f32 C holds (M = r + 8*(lane>=16), N = lane&15) ----
  const int rowHalf = (lane >> 4) * 8;
  v8f accs[4][4] = {{acc00, acc01, acc02, acc03},
                    {acc10, acc11, acc12, acc13},
                    {acc20, acc21, acc22, acc23},
                    {acc30, acc31, acc32, acc33}};
#pragma unroll
  for (int i = 0; i < 4; ++i) {
#pragma unroll
    for (int j = 0; j < 4; ++j) {
      size_t col = blockN + waveN * 64 + j * 16 + ln15;
      float bcol = bias ? bias[col] : 0.0f;
#pragma unroll
      for (int r = 0; r < 8; ++r) {
        size_t row = blockM + waveM * 64 + i * 16 + rowHalf + r;
        float v = accs[i][j][r] * alpha + bcol;
        if (act == 1) v = gelu_exact(v);
        if (row_scale) v *= row_scale[row * rs_stride];
        size_t idx = row * N + col;
        if (outF) { if (accum) outF[idx] += v; else outF[idx] = v; }
        if (outB) outB[idx] = f32_to_bf16(v);
      }
    }
  }
}

// ---------------- small kernels ----------------
__global__ void convert_bf16_kernel(const float* __restrict__ in, unsigned short* __restrict__ out, int n) {
  int i = blockIdx.x * blockDim.x + threadIdx.x;
  if (i < n) out[i] = f32_to_bf16(in[i]);
}
__global__ void zero_f32_kernel(float* __restrict__ p, int n) {
  int i = blockIdx.x * blockDim.x + threadIdx.x;
  if (i < n) p[i] = 0.0f;
}

// logits_all[t][z*E+e] = h[t,:] . zone_w[ze,:]  (wave per (t,ze))
__global__ void zone_logits_kernel(const float* __restrict__ h, const float* __restrict__ zone_w,
                                   float* __restrict__ logits) {
  int gw   = (blockIdx.x * blockDim.x + threadIdx.x) >> 5;
  int lane = threadIdx.x & 31;
  if (gw >= Tc * ZEc) return;
  int t = gw / ZEc, ze = gw % ZEc;
  const float* hr = h + (size_t)t * Dc;
  const float* wr = zone_w + (size_t)ze * Dc;
  float s = 0.0f;
  for (int i = lane; i < Dc; i += 32) s += hr[i] * wr[i];
  for (int o = 16; o; o >>= 1) s += __shfl_xor(s, o);
  if (lane == 0) logits[(size_t)t * ZEc + ze] = s;
}

// zone weights -> router softmax -> top2 -> composed_emb -> mix softmax (thread per token)
__global__ void router_kernel(const int* __restrict__ positions, const float* __restrict__ zb,
                              const float* __restrict__ ztemp, const float* __restrict__ zone_b,
                              const float* __restrict__ logits, const float* __restrict__ expert_emb,
                              const float* __restrict__ dec_w, const float* __restrict__ dec_b,
                              float* __restrict__ mix) {
  int t = blockIdx.x * blockDim.x + threadIdx.x;
  if (t >= Tc) return;
  int s = t % Sc;
  float pos = (float)positions[s];
  float temp = ztemp[0];
  float zw[Zc]; float zsum = 0.0f;
  for (int z = 0; z < Zc; ++z) {
    float a  = 1.0f / (1.0f + expf(-temp * (pos - zb[z])));
    float b2 = 1.0f / (1.0f + expf(-temp * (zb[z + 1] - pos)));
    zw[z] = a * b2; zsum += zw[z];
  }
  zsum = fmaxf(zsum, 1e-8f);
  float rl[Ec];
  for (int e = 0; e < Ec; ++e) {
    float v = 0.0f;
    for (int z = 0; z < Zc; ++z)
      v += (zw[z] / zsum) * (logits[(size_t)t * ZEc + z * Ec + e] + zone_b[z * Ec + e]);
    rl[e] = v;
  }
  float mx = rl[0];
  for (int e = 1; e < Ec; ++e) mx = fmaxf(mx, rl[e]);
  float sm = 0.0f, rw[Ec];
  for (int e = 0; e < Ec; ++e) { rw[e] = expf(rl[e] - mx); sm += rw[e]; }
  for (int e = 0; e < Ec; ++e) rw[e] /= sm;
  int i1 = 0;
  for (int e = 1; e < Ec; ++e) if (rw[e] > rw[i1]) i1 = e;
  int i2 = (i1 == 0) ? 1 : 0;
  for (int e = 0; e < Ec; ++e) if (e != i1 && rw[e] > rw[i2]) i2 = e;
  float v1 = rw[i1], v2 = rw[i2];
  float ml[Ec];
  for (int e = 0; e < Ec; ++e) ml[e] = dec_b[e];
  for (int c = 0; c < Cc; ++c) {
    float ce = v1 * expert_emb[i1 * Cc + c] + v2 * expert_emb[i2 * Cc + c];
    for (int e = 0; e < Ec; ++e) ml[e] += ce * dec_w[c * Ec + e];
  }
  float mx2 = ml[0];
  for (int e = 1; e < Ec; ++e) mx2 = fmaxf(mx2, ml[e]);
  float s2 = 0.0f;
  for (int e = 0; e < Ec; ++e) { ml[e] = expf(ml[e] - mx2); s2 += ml[e]; }
  for (int e = 0; e < Ec; ++e) mix[(size_t)t * Ec + e] = ml[e] / s2;
}

// softmax over M=128 memory slots (block of 128 per token) -> bf16 probs
__global__ void softmax128_kernel(const float* __restrict__ scores, unsigned short* __restrict__ attn) {
  __shared__ float red[128];
  int t = blockIdx.x, tid = threadIdx.x;
  float v = scores[(size_t)t * Mc + tid];
  red[tid] = v; __syncthreads();
  for (int o = 64; o; o >>= 1) { if (tid < o) red[tid] = fmaxf(red[tid], red[tid + o]); __syncthreads(); }
  float mx = red[0]; __syncthreads();
  float e = expf(v - mx);
  red[tid] = e; __syncthreads();
  for (int o = 64; o; o >>= 1) { if (tid < o) red[tid] += red[tid + o]; __syncthreads(); }
  float smv = red[0];
  attn[(size_t)t * Mc + tid] = f32_to_bf16(e / smv);
}

// per-token: sigmoid(gate_hidden . dg_w2 + dg_b2); atomic mean into diff_sum (wave per token)
__global__ void gate_reduce_kernel(const unsigned short* __restrict__ gate_bf,
                                   const float* __restrict__ dg_w2, const float* __restrict__ dg_b2,
                                   float* __restrict__ diff_sum) {
  int gw   = (blockIdx.x * blockDim.x + threadIdx.x) >> 5;
  int lane = threadIdx.x & 31;
  if (gw >= Tc) return;
  const unsigned short* g = gate_bf + (size_t)gw * DGc;
  float s = 0.0f;
  for (int i = lane; i < DGc; i += 32) s += bf16_to_f32(g[i]) * dg_w2[i];
  for (int o = 16; o; o >>= 1) s += __shfl_xor(s, o);
  if (lane == 0) {
    float sig = 1.0f / (1.0f + expf(-(s + dg_b2[0])));
    atomicAdd(diff_sum, sig * (1.0f / (float)Tc));
  }
}

// out = (diff > 0.5) ? a : b  (scalar device condition), dual f32/bf16 outputs
__global__ void select_kernel(const float* __restrict__ diff, const float* __restrict__ a,
                              const float* __restrict__ b, float* __restrict__ outF,
                              unsigned short* __restrict__ outB, int n) {
  int i = blockIdx.x * blockDim.x + threadIdx.x;
  if (i >= n) return;
  float v = (diff[0] > 0.5f) ? a[i] : b[i];
  outF[i] = v;
  if (outB) outB[i] = f32_to_bf16(v);
}

__global__ void final_add_kernel(const float* __restrict__ a, const float* __restrict__ b,
                                 const float* __restrict__ c, float* __restrict__ out, int n) {
  int i = blockIdx.x * blockDim.x + threadIdx.x;
  if (i < n) out[i] = a[i] + b[i] + c[i];
}

// ---------------- host orchestration ----------------
extern "C" void kernel_launch(void* const* d_in, const int* in_sizes, int n_in,
                              void* d_out, int out_size, void* d_ws, size_t ws_size,
                              hipStream_t stream) {
  const float* h          = (const float*)d_in[0];
  const int*   positions  = (const int*)d_in[1];
  const float* zb         = (const float*)d_in[2];
  const float* ztemp      = (const float*)d_in[3];
  const float* zone_w     = (const float*)d_in[4];
  const float* zone_b     = (const float*)d_in[5];
  const float* ew1        = (const float*)d_in[6];
  const float* eb1        = (const float*)d_in[7];
  const float* ew2        = (const float*)d_in[8];
  const float* eb2        = (const float*)d_in[9];
  const float* expert_emb = (const float*)d_in[10];
  const float* dec_w      = (const float*)d_in[11];
  const float* dec_b      = (const float*)d_in[12];
  const float* mem_keys   = (const float*)d_in[13];
  const float* mem_values = (const float*)d_in[14];
  const float* q_w        = (const float*)d_in[15];
  const float* q_b        = (const float*)d_in[16];
  const float* o_w        = (const float*)d_in[17];
  const float* o_b        = (const float*)d_in[18];
  const float* rc_w1      = (const float*)d_in[19];
  const float* rc_b1      = (const float*)d_in[20];
  const float* rc_w2      = (const float*)d_in[21];
  const float* rc_b2      = (const float*)d_in[22];
  const float* dg_w1      = (const float*)d_in[23];
  const float* dg_b1      = (const float*)d_in[24];
  const float* dg_w2      = (const float*)d_in[25];
  const float* dg_b2      = (const float*)d_in[26];
  const float* comb_w     = (const float*)d_in[27];
  const float* comb_b     = (const float*)d_in[28];
  float* out = (float*)d_out;

  // workspace carve-out
  char* w = (char*)d_ws;
  size_t off = 0;
  auto alloc = [&](size_t bytes) -> void* {
    void* p = w + off;
    off = (off + bytes + 255) & ~(size_t)255;
    return p;
  };
  typedef unsigned short us;
  us*    h_bf      = (us*)alloc((size_t)Tc * Dc * 2);
  us*    rcw1_bf   = (us*)alloc((size_t)Dc * D4c * 2);
  us*    rcw2_bf   = (us*)alloc((size_t)D4c * Dc * 2);
  us*    combw_bf  = (us*)alloc((size_t)2 * Dc * Dc * 2);
  us*    qw_bf     = (us*)alloc((size_t)Dc * Dc * 2);
  us*    ow_bf     = (us*)alloc((size_t)Dc * Dc * 2);
  us*    mk_bf     = (us*)alloc((size_t)Mc * Dc * 2);
  us*    mv_bf     = (us*)alloc((size_t)Mc * Dc * 2);
  us*    dgw1_bf   = (us*)alloc((size_t)Dc * DGc * 2);
  float* logits    = (float*)alloc((size_t)Tc * ZEc * 4);
  float* mix       = (float*)alloc((size_t)Tc * Ec * 4);
  us*    ew1_bf    = (us*)alloc((size_t)Dc * D4c * 2);   // per-expert staging
  us*    ew2_bf    = (us*)alloc((size_t)D4c * Dc * 2);
  us*    hmid_bf   = (us*)alloc((size_t)Tc * D4c * 2);   // reused: expert mid + rc mid
  float* composed  = (float*)alloc((size_t)Tc * Dc * 4);
  us*    q_bf      = (us*)alloc((size_t)Tc * Dc * 2);
  float* scores    = (float*)alloc((size_t)Tc * Mc * 4);
  us*    attn_bf   = (us*)alloc((size_t)Tc * Mc * 2);
  us*    mempre_bf = (us*)alloc((size_t)Tc * Dc * 2);
  float* memout    = (float*)alloc((size_t)Tc * Dc * 4);
  float* cf[4]; us* cb[4];
  for (int k = 0; k < 4; ++k) {
    cf[k] = (float*)alloc((size_t)Tc * Dc * 4);
    cb[k] = (us*)alloc((size_t)Tc * Dc * 2);
  }
  us*    gate_bf   = (us*)alloc((size_t)Tc * DGc * 2);
  float* diffs     = (float*)alloc(4 * 4);               // diff0..diff2 sums
  float* comb_out  = (float*)alloc((size_t)Tc * Dc * 4);
  float* r2f = (float*)alloc((size_t)Tc * Dc * 4); us* r2b = (us*)alloc((size_t)Tc * Dc * 2);
  float* r1f = (float*)alloc((size_t)Tc * Dc * 4); us* r1b = (us*)alloc((size_t)Tc * Dc * 2);
  float* recf = (float*)alloc((size_t)Tc * Dc * 4);
  (void)ws_size; (void)n_in; (void)in_sizes; (void)out_size;

  auto conv = [&](const float* src, us* dst, int n) {
    convert_bf16_kernel<<<(n + 255) / 256, 256, 0, stream>>>(src, dst, n);
  };
  auto zero = [&](float* p, int n) {
    zero_f32_kernel<<<(n + 255) / 256, 256, 0, stream>>>(p, n);
  };
  auto gemm = [&](const us* A, const us* Bm, int M, int N, int K, int btr,
                  const float* bias, float alpha, int act,
                  const float* rscale, int rstride, int accum, float* oF, us* oB) {
    dim3 grid(N / TN, M / TM);
    if (btr)
      gemm_bf16_wmma<1><<<grid, 256, 0, stream>>>(A, Bm, M, N, K, bias, alpha, act,
                                                  rscale, rstride, accum, oF, oB);
    else
      gemm_bf16_wmma<0><<<grid, 256, 0, stream>>>(A, Bm, M, N, K, bias, alpha, act,
                                                  rscale, rstride, accum, oF, oB);
  };

  // 0) init accumulators
  zero(composed, Tc * Dc);
  zero(diffs, 4);

  // 1) bf16 conversions of activations + persistent weights
  conv(h, h_bf, Tc * Dc);
  conv(rc_w1, rcw1_bf, Dc * D4c);
  conv(rc_w2, rcw2_bf, D4c * Dc);
  conv(comb_w, combw_bf, 2 * Dc * Dc);
  conv(q_w, qw_bf, Dc * Dc);
  conv(o_w, ow_bf, Dc * Dc);
  conv(mem_keys, mk_bf, Mc * Dc);
  conv(mem_values, mv_bf, Mc * Dc);
  conv(dg_w1, dgw1_bf, Dc * DGc);

  // 2) router path
  zone_logits_kernel<<<(Tc * ZEc) / 8, 256, 0, stream>>>(h, zone_w, logits);
  router_kernel<<<(Tc + 255) / 256, 256, 0, stream>>>(positions, zb, ztemp, zone_b, logits,
                                                      expert_emb, dec_w, dec_b, mix);

  // 3) dense experts, mixed directly into `composed`
  for (int e = 0; e < Ec; ++e) {
    conv(ew1 + (size_t)e * Dc * D4c, ew1_bf, Dc * D4c);
    conv(ew2 + (size_t)e * D4c * Dc, ew2_bf, D4c * Dc);
    gemm(h_bf, ew1_bf, Tc, D4c, Dc, 0, eb1 + (size_t)e * D4c, 1.0f, /*gelu*/1,
         nullptr, 0, 0, nullptr, hmid_bf);
    gemm(hmid_bf, ew2_bf, Tc, Dc, D4c, 0, eb2 + (size_t)e * Dc, 1.0f, 0,
         mix + e, Ec, /*accum*/1, composed, nullptr);
  }

  // 4) memory attention
  gemm(h_bf, qw_bf, Tc, Dc, Dc, 0, q_b, 1.0f, 0, nullptr, 0, 0, nullptr, q_bf);
  gemm(q_bf, mk_bf, Tc, Mc, Dc, /*b_trans*/1, nullptr, 0.0625f /*1/sqrt(256)*/, 0,
       nullptr, 0, 0, scores, nullptr);
  softmax128_kernel<<<Tc, 128, 0, stream>>>(scores, attn_bf);
  gemm(attn_bf, mv_bf, Tc, Dc, Mc, 0, nullptr, 1.0f, 0, nullptr, 0, 0, nullptr, mempre_bf);
  gemm(mempre_bf, ow_bf, Tc, Dc, Dc, 0, o_b, 1.0f, 0, nullptr, 0, 0, memout, nullptr);

  // 5) recursive refinement: c0..c3 = iterated rc_mlp
  const us* rc_in = h_bf;
  for (int k = 0; k < 4; ++k) {
    gemm(rc_in, rcw1_bf, Tc, D4c, Dc, 0, rc_b1, 1.0f, /*gelu*/1, nullptr, 0, 0, nullptr, hmid_bf);
    gemm(hmid_bf, rcw2_bf, Tc, Dc, D4c, 0, rc_b2, 1.0f, 0, nullptr, 0, 0, cf[k], cb[k]);
    rc_in = cb[k];
  }
  // gates on c0,c1,c2 -> scalar means diffs[0..2]
  for (int k = 0; k < 3; ++k) {
    gemm(cb[k], dgw1_bf, Tc, DGc, Dc, 0, dg_b1, 1.0f, /*gelu*/1, nullptr, 0, 0, nullptr, gate_bf);
    gate_reduce_kernel<<<Tc / 8, 256, 0, stream>>>(gate_bf, dg_w2, dg_b2, diffs + k);
  }
  const us* combw_top = combw_bf;                       // rows 0..D-1
  const us* combw_bot = combw_bf + (size_t)Dc * Dc;     // rows D..2D-1
  const int nTD = Tc * Dc;
  // depth 2: out2 = [c2,c3]@comb_w + comb_b ; r2 = diff2>0.5 ? out2 : c2
  gemm(cb[2], combw_top, Tc, Dc, Dc, 0, comb_b, 1.0f, 0, nullptr, 0, 0, comb_out, nullptr);
  gemm(cb[3], combw_bot, Tc, Dc, Dc, 0, nullptr, 1.0f, 0, nullptr, 0, 1, comb_out, nullptr);
  select_kernel<<<(nTD + 255) / 256, 256, 0, stream>>>(diffs + 2, comb_out, cf[2], r2f, r2b, nTD);
  // depth 1
  gemm(cb[1], combw_top, Tc, Dc, Dc, 0, comb_b, 1.0f, 0, nullptr, 0, 0, comb_out, nullptr);
  gemm(r2b, combw_bot, Tc, Dc, Dc, 0, nullptr, 1.0f, 0, nullptr, 0, 1, comb_out, nullptr);
  select_kernel<<<(nTD + 255) / 256, 256, 0, stream>>>(diffs + 1, comb_out, cf[1], r1f, r1b, nTD);
  // depth 0
  gemm(cb[0], combw_top, Tc, Dc, Dc, 0, comb_b, 1.0f, 0, nullptr, 0, 0, comb_out, nullptr);
  gemm(r1b, combw_bot, Tc, Dc, Dc, 0, nullptr, 1.0f, 0, nullptr, 0, 1, comb_out, nullptr);
  select_kernel<<<(nTD + 255) / 256, 256, 0, stream>>>(diffs + 0, comb_out, cf[0], recf, nullptr, nTD);

  // 6) final sum
  final_add_kernel<<<(nTD + 255) / 256, 256, 0, stream>>>(composed, memout, recf, out, nTD);
}
